// TemporalModel_35716948033802
// MI455X (gfx1250) — compile-verified, matrix operand order checked by reading, not run on hardware
//
#include <hip/hip_runtime.h>
#include <hip/hip_bf16.h>
#include <math.h>

typedef __attribute__((ext_vector_type(16))) _Float16 v16h;
typedef __attribute__((ext_vector_type(8)))  float    v8f;

#define D_MODEL 256
#define D_INNER 512
#define D_STATE 16
#define DT_RANK 16
#define SEQ     4096      // tokens per batch (NSEG*SEGL)
#define NTOK    8192      // BATCH * SEQ
#define NDIR    4         // DEPTH * 2 directions

__device__ __forceinline__ float sigm_(float x){ return 1.f / (1.f + __expf(-x)); }
__device__ __forceinline__ float silu_(float x){ return x * sigm_(x); }

// 32-bit LDS byte offset of a __shared__ object (generic -> addrspace(3) -> int)
typedef __attribute__((address_space(3))) char lds_char;
__device__ __forceinline__ unsigned lds_off(const void* p){
  return (unsigned)(unsigned long long)(lds_char*)(p);
}
// CDNA5 async copy: 16 bytes global -> LDS, tracked by ASYNCcnt
#define ASYNC_CP_B128(ldsoff, gptr) \
  asm volatile("global_load_async_to_lds_b128 %0, %1, off" :: "v"(ldsoff), "v"(gptr) : "memory")
#define WAIT_ASYNC0() asm volatile("s_wait_asynccnt 0" ::: "memory")

// ---------------------------------------------------------------- weight prep
__global__ void k_cast_f16(const float* __restrict__ src, _Float16* __restrict__ dst, int n){
  int i = blockIdx.x * blockDim.x + threadIdx.x;
  if (i < n) dst[i] = (_Float16)src[i];
}
__global__ void k_negexp(const float* __restrict__ src, float* __restrict__ dst, int n){
  int i = blockIdx.x * blockDim.x + threadIdx.x;
  if (i < n) dst[i] = -__expf(src[i]);
}

// ---------------------------------------------------------------- WMMA GEMM
// C[M,N] = rowmap(A[M,K] f16) @ W[N,K]^T (+bias)(+seg/pos embeddings)
// rowmap = per-batch time flip of A rows when flip!=0 (backward Mamba dir).
// 256 threads = 8 waves; each wave owns 16 M-rows x 64 N-cols (4 WMMA accums).
// Double-buffered LDS filled by global_load_async_to_lds_b128, pipeline
// unrolled x2 so buffer indices are compile-time constants.
#define BM 128
#define BN 64
#define BK 32
#define LDS_P 40   // half stride: 80B rows -> b128-aligned, conflict-free frags

// one K-stage: preload A frag + 4 B frags, then 4 back-to-back WMMAs
__device__ __forceinline__ void gemm_stage(
    const _Float16 (*__restrict__ As)[LDS_P], const _Float16 (*__restrict__ Ws)[LDS_P],
    int mrow, int khi, int lane, v8f (&acc)[4])
{
  v16h af;
  #pragma unroll
  for (int v = 0; v < 8; ++v) {
    int k = ((v & 3) << 1) + ((v >> 2) << 4) + khi;
    af[2 * v]     = As[mrow][k];
    af[2 * v + 1] = As[mrow][k + 1];
  }
  v16h bf[4];
  #pragma unroll
  for (int j = 0; j < 4; ++j) {
    int ncol = (j << 4) + (lane & 15);
    #pragma unroll
    for (int v = 0; v < 8; ++v) {
      int k = ((v & 3) << 1) + ((v >> 2) << 4) + khi;
      bf[j][2 * v]     = Ws[ncol][k];
      bf[j][2 * v + 1] = Ws[ncol][k + 1];
    }
  }
  #pragma unroll
  for (int j = 0; j < 4; ++j)
    acc[j] = __builtin_amdgcn_wmma_f32_16x16x32_f16(
        false, af, false, bf[j], (short)0, acc[j], false, false);
}

__global__ __launch_bounds__(256) void k_gemm(
    const _Float16* __restrict__ Af, const _Float16* __restrict__ Wf,
    const float* __restrict__ bias, float* __restrict__ C,
    int M, int N, int K, int flip,
    const float* __restrict__ seg, const float* __restrict__ pos)
{
  __shared__ _Float16 As[2][BM][LDS_P];
  __shared__ _Float16 Ws[2][BN][LDS_P];

  const int tid  = threadIdx.x;
  const int lane = tid & 31;
  const int wave = tid >> 5;
  const int m0   = blockIdx.y * BM;
  const int n0   = blockIdx.x * BN;

  // per-thread fill descriptors, fixed across the K loop
  const _Float16* pA[2];
  unsigned lA[2][2];                       // [buffer][chunk]
  #pragma unroll
  for (int it = 0; it < 2; ++it) {
    int idx8 = (tid + it * 256) * 8;       // 8 halves per b128
    int r = idx8 >> 5, c = idx8 & 31;
    int gr = m0 + r, src = gr;
    if (flip) { int b = gr >> 12, p = gr & (SEQ - 1); src = (b << 12) + (SEQ - 1 - p); }
    pA[it] = Af + (size_t)src * K + c;
    lA[0][it] = lds_off(&As[0][r][c]);
    lA[1][it] = lds_off(&As[1][r][c]);
  }
  const int idx8w = tid * 8;
  const int rW = idx8w >> 5, cW = idx8w & 31;
  const int gn = n0 + rW;
  const bool wv = gn < N;
  const _Float16* pW = Wf + (size_t)(wv ? gn : 0) * K + cW;
  const unsigned lW[2] = { lds_off(&Ws[0][rW][cW]), lds_off(&Ws[1][rW][cW]) };
  if (!wv) {                                // zero-pad columns past N (x_proj N=48)
    #pragma unroll
    for (int c = 0; c < 8; ++c) { Ws[0][rW][cW + c] = (_Float16)0.f; Ws[1][rW][cW + c] = (_Float16)0.f; }
  }

  v8f acc[4] = {};
  const int nsteps = K / BK;
  const int mrow = (wave << 4) + (lane & 15);
  const int khi  = (lane >> 4) << 3;

  // fill issues advance the global pointers by BK halves each call
  auto fill = [&](unsigned la0, unsigned la1, unsigned lw) {
    ASYNC_CP_B128(la0, pA[0]);
    ASYNC_CP_B128(la1, pA[1]);
    if (wv) ASYNC_CP_B128(lw, pW);
    pA[0] += BK; pA[1] += BK; pW += BK;
  };

  fill(lA[0][0], lA[0][1], lW[0]);          // prologue: buffer 0, k=0
  int st = 0;
  for (;;) {
    // ---- stage: buffer 0
    WAIT_ASYNC0();
    __syncthreads();
    if (st + 1 < nsteps) fill(lA[1][0], lA[1][1], lW[1]);
    gemm_stage(As[0], Ws[0], mrow, khi, lane, acc);
    if (++st == nsteps) break;
    // ---- stage: buffer 1
    WAIT_ASYNC0();
    __syncthreads();
    if (st + 1 < nsteps) fill(lA[0][0], lA[0][1], lW[0]);
    gemm_stage(As[1], Ws[1], mrow, khi, lane, acc);
    if (++st == nsteps) break;
  }

  // Epilogue. D layout: VGPR r -> M = r + 8*(lane>=16), N = lane&15
  const int mbase = m0 + (wave << 4) + ((lane >> 4) << 3);
  const int nn = lane & 15;
  #pragma unroll
  for (int j = 0; j < 4; ++j) {
    int col = n0 + (j << 4) + nn;
    if (col < N) {
      float badd = bias ? bias[col] : 0.f;
      #pragma unroll
      for (int r = 0; r < 8; ++r) {
        int row = mbase + r;
        float v = acc[j][r] + badd;
        if (seg) {  // proj kernel: add segment + positional embeddings
          int p = row & (SEQ - 1);
          v += seg[(p >> 9) * D_MODEL + col] + pos[(p & 511) * D_MODEL + col];
        }
        C[(long)row * N + col] = v;
      }
    }
  }
}

// ---------------------------------------------------------------- layernorm
__global__ __launch_bounds__(256) void k_layernorm(
    const float* __restrict__ h, const float* __restrict__ g,
    const float* __restrict__ b, _Float16* __restrict__ xnh)
{
  __shared__ float red[256];
  int t = blockIdx.x, c = threadIdx.x;
  float v = h[(long)t * D_MODEL + c];
  red[c] = v; __syncthreads();
  for (int s = 128; s > 0; s >>= 1) { if (c < s) red[c] += red[c + s]; __syncthreads(); }
  float mu = red[0] * (1.f / D_MODEL);
  __syncthreads();
  float d = v - mu;
  red[c] = d * d; __syncthreads();
  for (int s = 128; s > 0; s >>= 1) { if (c < s) red[c] += red[c + s]; __syncthreads(); }
  float r = rsqrtf(red[0] * (1.f / D_MODEL) + 1e-5f);
  xnh[(long)t * D_MODEL + c] = (_Float16)(d * r * g[c] + b[c]);
}

// ----------------------------------------------- causal depthwise conv + SiLU
__global__ void k_conv(const float* __restrict__ xz, const float* __restrict__ cw,
                       const float* __restrict__ cb, float* __restrict__ xc,
                       _Float16* __restrict__ xch, int n)
{
  int i = blockIdx.x * blockDim.x + threadIdx.x;
  if (i >= n) return;
  int tok = i >> 9, d = i & 511;
  int p = tok & (SEQ - 1);
  float acc = cb[d];
  #pragma unroll
  for (int k = 0; k < 4; ++k) {
    int pp = p - 3 + k;
    if (pp >= 0) acc = fmaf(xz[(long)(tok - 3 + k) * 1024 + d], cw[d * 4 + k], acc);
  }
  float v = silu_(acc);
  xc[i] = v;
  xch[i] = (_Float16)v;
}

// ---------------------------------------------------------- dt-proj + softplus
__global__ void k_dtproj(const float* __restrict__ dbl, const float* __restrict__ dtw,
                         const float* __restrict__ dtb, float* __restrict__ dt, int n)
{
  int i = blockIdx.x * blockDim.x + threadIdx.x;
  if (i >= n) return;
  int tok = i >> 9, d = i & 511;
  const float* row = dbl + (long)tok * 48;
  const float* w   = dtw + d * DT_RANK;
  float acc = dtb[d];
  #pragma unroll
  for (int k = 0; k < DT_RANK; ++k) acc = fmaf(row[k], w[k], acc);
  dt[i] = (acc > 20.f) ? acc : log1pf(__expf(acc));
}

// ------------------------------------------------------------- selective scan
// One thread per (b, d, s); lane bits[3:0]=s so y-reduction is a 16-lane xor tree.
__global__ __launch_bounds__(256) void k_scan(
    const float* __restrict__ dt, const float* __restrict__ xc,
    const float* __restrict__ dbl, const float* __restrict__ An,
    float* __restrict__ ys)
{
  int gid = blockIdx.x * 256 + threadIdx.x;          // 16384 threads
  int s = gid & 15;
  int d = (gid >> 4) & 511;
  int b = gid >> 13;
  float A = An[d * 16 + s];
  float h = 0.f;
  long tokbase = (long)b * SEQ;
  for (int t = 0; t < SEQ; ++t) {
    long tok = tokbase + t;
    float dtv = dt[tok * 512 + d];
    float xv  = xc[tok * 512 + d];
    float Bv  = dbl[tok * 48 + 16 + s];
    float Cv  = dbl[tok * 48 + 32 + s];
    h = __expf(dtv * A) * h + (dtv * xv) * Bv;       // critical chain: exp+fma
    float p = h * Cv;                                // reduction off-chain
    p += __shfl_xor(p, 8, 32);
    p += __shfl_xor(p, 4, 32);
    p += __shfl_xor(p, 2, 32);
    p += __shfl_xor(p, 1, 32);
    if (s == 0) ys[tok * 512 + d] = p;
  }
}

// ------------------------------------------------------- D-skip + SiLU gating
__global__ void k_gate(const float* __restrict__ ys, const float* __restrict__ xc,
                       const float* __restrict__ xz, const float* __restrict__ Dp,
                       _Float16* __restrict__ ysh, int n)
{
  int i = blockIdx.x * blockDim.x + threadIdx.x;
  if (i >= n) return;
  int tok = i >> 9, d = i & 511;
  float z = xz[(long)tok * 1024 + 512 + d];
  ysh[i] = (_Float16)((ys[i] + xc[i] * Dp[d]) * silu_(z));
}

// -------------------------------------- residual merge: h += w*yf + (1-w)*flip(yb)
__global__ void k_accum(float* __restrict__ h, const float* __restrict__ yf,
                        const float* __restrict__ yb, const float* __restrict__ alpha, int n)
{
  int i = blockIdx.x * blockDim.x + threadIdx.x;
  if (i >= n) return;
  float w = sigm_(alpha[0]);
  int tok = i >> 8, c = i & 255;
  int b = tok >> 12, p = tok & (SEQ - 1);
  long jb = ((long)(b << 12) + (SEQ - 1 - p)) * 256 + c;
  h[i] += w * yf[i] + (1.f - w) * yb[jb];
}

// ================================================================= launcher
extern "C" void kernel_launch(void* const* d_in, const int* in_sizes, int n_in,
                              void* d_out, int out_size, void* d_ws, size_t ws_size,
                              hipStream_t stream)
{
  const float* x      = (const float*)d_in[0];
  const float* proj_w = (const float*)d_in[1];
  const float* proj_b = (const float*)d_in[2];
  const float* seg_e  = (const float*)d_in[3];
  const float* pos_e  = (const float*)d_in[4];
  const float* ln_g   = (const float*)d_in[5];
  const float* ln_b   = (const float*)d_in[6];
  const float* alpha  = (const float*)d_in[7];
  const float* in_w   = (const float*)d_in[8];
  const float* conv_w = (const float*)d_in[9];
  const float* conv_b = (const float*)d_in[10];
  const float* xp_w   = (const float*)d_in[11];
  const float* dt_w   = (const float*)d_in[12];
  const float* dt_b   = (const float*)d_in[13];
  const float* A_log  = (const float*)d_in[14];
  const float* Dp     = (const float*)d_in[15];
  const float* out_w  = (const float*)d_in[16];
  const float* head_w = (const float*)d_in[17];
  const float* head_b = (const float*)d_in[18];

  char* ws = (char*)d_ws;
  size_t off = 0;
  auto alloc = [&](size_t bytes) -> void* {
    void* p = ws + off; off += (bytes + 255) & ~(size_t)255; return p;
  };
  float*    h    = (float*)   alloc((size_t)NTOK * 256 * 4);
  float*    xz   = (float*)   alloc((size_t)NTOK * 1024 * 4);
  float*    xc   = (float*)   alloc((size_t)NTOK * 512 * 4);
  float*    dbl  = (float*)   alloc((size_t)NTOK * 48 * 4);
  float*    dtb  = (float*)   alloc((size_t)NTOK * 512 * 4);
  float*    ysb  = (float*)   alloc((size_t)NTOK * 512 * 4);
  float*    yf   = (float*)   alloc((size_t)NTOK * 256 * 4);
  float*    yb   = (float*)   alloc((size_t)NTOK * 256 * 4);
  _Float16* xh   = (_Float16*)alloc((size_t)NTOK * 256 * 2);
  _Float16* xnh  = (_Float16*)alloc((size_t)NTOK * 256 * 2);
  _Float16* xch  = (_Float16*)alloc((size_t)NTOK * 512 * 2);
  _Float16* ysh  = (_Float16*)alloc((size_t)NTOK * 512 * 2);
  _Float16* hh   = (_Float16*)alloc((size_t)NTOK * 256 * 2);
  _Float16* prjh = (_Float16*)alloc((size_t)256 * 256 * 2);
  _Float16* hedh = (_Float16*)alloc((size_t)256 * 256 * 2);
  _Float16* inh  = (_Float16*)alloc((size_t)NDIR * 1024 * 256 * 2);
  _Float16* xph  = (_Float16*)alloc((size_t)NDIR * 48 * 512 * 2);
  _Float16* outh = (_Float16*)alloc((size_t)NDIR * 256 * 512 * 2);
  float*    Aneg = (float*)   alloc((size_t)NDIR * 512 * 16 * 4);

  auto cast = [&](const float* s, _Float16* d, int n){
    k_cast_f16<<<(n + 255) / 256, 256, 0, stream>>>(s, d, n);
  };
  cast(proj_w, prjh, 256 * 256);
  cast(head_w, hedh, 256 * 256);
  cast(in_w,   inh,  NDIR * 1024 * 256);
  cast(xp_w,   xph,  NDIR * 48 * 512);
  cast(out_w,  outh, NDIR * 256 * 512);
  cast(x,      xh,   NTOK * 256);
  k_negexp<<<(NDIR * 512 * 16 + 255) / 256, 256, 0, stream>>>(A_log, Aneg, NDIR * 512 * 16);

  auto gemm = [&](const _Float16* A, const _Float16* W, const float* bias, float* C,
                  int M, int N, int K, int flip, const float* sg, const float* ps){
    dim3 grid((N + BN - 1) / BN, M / BM);
    k_gemm<<<grid, 256, 0, stream>>>(A, W, bias, C, M, N, K, flip, sg, ps);
  };

  // projection + embeddings
  gemm(xh, prjh, proj_b, h, NTOK, 256, 256, 0, seg_e, pos_e);

  const int nE512 = NTOK * 512, nE256 = NTOK * 256;
  for (int layer = 0; layer < 2; ++layer) {
    k_layernorm<<<NTOK, 256, 0, stream>>>(h, ln_g + layer * 256, ln_b + layer * 256, xnh);
    for (int dir = 0; dir < 2; ++dir) {
      int wi = layer * 2 + dir;
      gemm(xnh, inh + (size_t)wi * 1024 * 256, nullptr, xz, NTOK, 1024, 256, dir, nullptr, nullptr);
      k_conv<<<nE512 / 256, 256, 0, stream>>>(xz, conv_w + wi * 512 * 4, conv_b + wi * 512, xc, xch, nE512);
      gemm(xch, xph + (size_t)wi * 48 * 512, nullptr, dbl, NTOK, 48, 512, 0, nullptr, nullptr);
      k_dtproj<<<nE512 / 256, 256, 0, stream>>>(dbl, dt_w + wi * 512 * 16, dt_b + wi * 512, dtb, nE512);
      k_scan<<<64, 256, 0, stream>>>(dtb, xc, dbl, Aneg + wi * 512 * 16, ysb);
      k_gate<<<nE512 / 256, 256, 0, stream>>>(ysb, xc, xz, Dp + wi * 512, ysh, nE512);
      gemm(ysh, outh + (size_t)wi * 256 * 512, nullptr, dir ? yb : yf, NTOK, 256, 512, 0, nullptr, nullptr);
    }
    k_accum<<<nE256 / 256, 256, 0, stream>>>(h, yf, yb, alpha + layer, nE256);
  }

  // head
  k_cast_f16<<<nE256 / 256, 256, 0, stream>>>(h, hh, nE256);
  gemm(hh, hedh, head_b, (float*)d_out, NTOK, 256, 256, 0, nullptr, nullptr);
  (void)in_sizes; (void)n_in; (void)out_size; (void)ws_size;
}
